// UnitsHead_54906861912300
// MI455X (gfx1250) — matrix operand order repaired and checked by reading, not compiled, not run on hardware
//
#include <hip/hip_runtime.h>
#include <hip/hip_bf16.h>
#include <math.h>

// ---------------- problem constants ----------------
#define BB   512   // batch
#define EE   512   // max entities
#define LL   1024  // latent
#define DD   256   // ent dim
#define HH   512   // hidden
#define SS   13    // steps
#define EOFI 512   // eof index
#define NEGV (-1000000000.0f)

typedef __attribute__((ext_vector_type(16))) _Float16 v16h;
typedef __attribute__((ext_vector_type(8)))  _Float16 v8h;
typedef __attribute__((ext_vector_type(8)))  float    v8f;

// ---------------- output layout (flat f32, return order) ----------------
static constexpr size_t LOGITS_N = (size_t)BB * SS * (EE + 1);
static constexpr size_t IDS_OFF  = LOGITS_N;
static constexpr size_t SEL_OFF  = IDS_OFF + (size_t)BB * SS;
static constexpr size_t SUM_OFF  = SEL_OFF + (size_t)BB * SS;

// ---------------- workspace layout (bytes) ----------------
// Packed-to-fragment weight copies (f16), activation f16 copies, f32 state.
static constexpr size_t OFF_W2P   = 0;                         // [1024*2048] f16, frag-packed
static constexpr size_t OFF_WC1P  = OFF_W2P   + 1024*2048*2;   // [1024*1536] f16, frag-packed
static constexpr size_t OFF_BQ2P  = OFF_WC1P  + 1024*1536*2;   // [512*512]   f16, frag-packed
static constexpr size_t OFF_EPWP  = OFF_BQ2P  + 512*512*2;     // [512*256]   f16, frag-packed (ep_w^T)
static constexpr size_t OFF_LATH  = OFF_EPWP  + 512*256*2;     // [512][1024] f16 row-major
static constexpr size_t OFF_A2H   = OFF_LATH  + 512*1024*2;    // [512][1024] f16 (x | hidden)
static constexpr size_t OFF_GH    = OFF_A2H   + 512*1024*2;    // [512][512]  f16 (gelu out)
static constexpr size_t OFF_HIDH  = OFF_GH    + 512*512*2;     // [512][512]  f16
static constexpr size_t OFF_T1    = OFF_HIDH  + 512*512*2;     // [512][1536] f32
static constexpr size_t OFF_GATES = OFF_T1    + (size_t)512*1536*4; // [512][2048] f32
static constexpr size_t OFF_BQ    = OFF_GATES + (size_t)512*2048*4; // [512][512] f32
static constexpr size_t OFF_HIDF  = OFF_BQ    + 512*512*4;     // [512][512] f32
static constexpr size_t OFF_CELL  = OFF_HIDF  + 512*512*4;     // [512][512] f32
static constexpr size_t OFF_Q     = OFF_CELL  + 512*512*4;     // [512][256] f32
static constexpr size_t OFF_REM   = OFF_Q     + 512*256*4;     // [512][512] i32
static constexpr size_t OFF_FIN   = OFF_REM   + 512*512*4;     // [512] i32
static constexpr size_t OFF_CHO   = OFF_FIN   + 512*4;         // [512] i32
static constexpr size_t OFF_IDS   = OFF_CHO   + 512*4;         // [512][13] i32
static constexpr size_t OFF_SEL   = OFF_IDS   + 512*13*4;      // [512][13] i32

__device__ __forceinline__ float sigf(float x) { return 1.0f / (1.0f + __expf(-x)); }

// Map a flat index in the frag-packed B buffer to (k,n) of the logical B[K,N].
// Fragment layout (wave32, v_wmma_*_16x16x32_f16 B operand):
//   lane l (n = nt*16 + (l&15)) holds 16 contiguous halves = B[kt*32 + (l>>4)*16 + j][n]
__device__ __forceinline__ void bfrag_kn(int i, int N, int& k, int& n) {
  int j    = i & 15;
  int l    = (i >> 4) & 31;
  int tile = i >> 9;
  int ntiles = N >> 4;
  int nt = tile % ntiles;
  int kt = tile / ntiles;
  k = kt * 32 + ((l >> 4) & 1) * 16 + j;
  n = nt * 16 + (l & 15);
}

// ======================= weight packing (f32 -> frag-packed f16) ===============
__global__ __launch_bounds__(256) void k_pack_w2_frag(const float* __restrict__ w_ih,
                                                      const float* __restrict__ w_hh,
                                                      _Float16* __restrict__ P) {
  const int n_tot = 1024 * 2048;
  for (int i = blockIdx.x * 256 + threadIdx.x; i < n_tot; i += gridDim.x * 256) {
    int k, n; bfrag_kn(i, 2048, k, n);
    float v = (k < 512) ? w_ih[(size_t)k * 2048 + n] : w_hh[(size_t)(k - 512) * 2048 + n];
    P[i] = (_Float16)v;
  }
}

__global__ __launch_bounds__(256) void k_pack_wc1_frag(const float* __restrict__ bq_w1,
                                                       const float* __restrict__ hi_w,
                                                       const float* __restrict__ ci_w,
                                                       _Float16* __restrict__ P) {
  const int n_tot = 1024 * 1536;
  for (int i = blockIdx.x * 256 + threadIdx.x; i < n_tot; i += gridDim.x * 256) {
    int k, n; bfrag_kn(i, 1536, k, n);
    float v;
    if (n < 512)       v = bq_w1[(size_t)k * 512 + n];
    else if (n < 1024) v = hi_w[(size_t)k * 512 + (n - 512)];
    else               v = ci_w[(size_t)k * 512 + (n - 1024)];
    P[i] = (_Float16)v;
  }
}

__global__ __launch_bounds__(256) void k_pack_bq2_frag(const float* __restrict__ bq_w2,
                                                       _Float16* __restrict__ P) {
  const int n_tot = 512 * 512;
  for (int i = blockIdx.x * 256 + threadIdx.x; i < n_tot; i += gridDim.x * 256) {
    int k, n; bfrag_kn(i, 512, k, n);
    P[i] = (_Float16)bq_w2[(size_t)k * 512 + n];
  }
}

// B = ep_w^T : B[k][n] = ep_w[n][k],  K=512 (hidden), N=256 (ent dim)
__global__ __launch_bounds__(256) void k_pack_epwt_frag(const float* __restrict__ ep_w,
                                                        _Float16* __restrict__ P) {
  const int n_tot = 512 * 256;
  for (int i = blockIdx.x * 256 + threadIdx.x; i < n_tot; i += gridDim.x * 256) {
    int k, n; bfrag_kn(i, 256, k, n);
    P[i] = (_Float16)ep_w[(size_t)n * 512 + k];
  }
}

__global__ __launch_bounds__(256) void k_cvt_f16(const float* __restrict__ s,
                                                 _Float16* __restrict__ d, int n) {
  for (int i = blockIdx.x * 256 + threadIdx.x; i < n; i += gridDim.x * 256)
    d[i] = (_Float16)s[i];
}

__global__ __launch_bounds__(256) void k_init_state(const int* __restrict__ mask,
                                                    int* __restrict__ REM,
                                                    int* __restrict__ FIN) {
  int i = blockIdx.x * 256 + threadIdx.x;   // over B*E
  if (i < BB * EE) {
    REM[i] = mask[i] ? 1 : 0;
    if ((i & (EE - 1)) == 0) FIN[i >> 9] = 0;
  }
}

// ======================= LDS-free WMMA GEMM ====================================
// C[512,N] = A[512,K] @ B[K,N] (+bias).  A: row-major f16 (lda halves).
// Bp: frag-packed f16 (see bfrag_kn).  Block = 4 waves = 64x128 tile.
// Each wave: 32x64 tile -> 8 x v_wmma_f32_16x16x32_f16 per 32-wide K step.
// A fragment = two contiguous 16B global loads per lane; B fragment = one
// contiguous 32B per lane (coalesced 1KB/wave), all L2-resident.
__global__ __launch_bounds__(128) void k_gemm_wmma(const _Float16* __restrict__ A,
                                                   const _Float16* __restrict__ Bp,
                                                   const float* __restrict__ bias,
                                                   float* __restrict__ C,
                                                   int N, int K, int lda, int ldc) {
  const int lane  = threadIdx.x & 31;
  const int wave  = threadIdx.x >> 5;
  const int wr    = (wave >> 1) * 32;          // wave row offset inside 64-row tile
  const int wnt   = (wave & 1) * 4;            // wave's first n-tile (4 tiles = 64 cols)
  const int brow  = blockIdx.y * 64;
  const int btile = blockIdx.x * 8;            // 8 n-tiles per block (128 cols)
  const int ntiles = N >> 4;
  const int mrow  = lane & 15;
  const int hi    = (lane >> 4) & 1;

  v8f acc[2][4] = {};

  for (int kk = 0; kk < K; kk += 32) {
    // ---- A fragments (two 16-row sub-tiles) ----
    v16h a[2];
#pragma unroll
    for (int ms = 0; ms < 2; ++ms) {
      const _Float16* ap = A + (size_t)(brow + wr + ms * 16 + mrow) * lda + kk + hi * 16;
      v8h lo = *(const v8h*)(ap);        // k = kk + hi*16 + 0..7
      v8h hv = *(const v8h*)(ap + 8);    // k = kk + hi*16 + 8..15
#pragma unroll
      for (int j = 0; j < 8; ++j) { a[ms][j] = lo[j]; a[ms][8 + j] = hv[j]; }
    }
    // ---- B fragments + 8 WMMAs ----
    const _Float16* bbase =
        Bp + (((size_t)(kk >> 5) * ntiles + (btile + wnt)) * 32 + lane) * 16;
#pragma unroll
    for (int nt = 0; nt < 4; ++nt) {
      v16h bf = *(const v16h*)(bbase + (size_t)nt * 32 * 16);
      acc[0][nt] = __builtin_amdgcn_wmma_f32_16x16x32_f16(false, a[0], false, bf,
                                                          (short)0, acc[0][nt], false, false);
      acc[1][nt] = __builtin_amdgcn_wmma_f32_16x16x32_f16(false, a[1], false, bf,
                                                          (short)0, acc[1][nt], false, false);
    }
  }

#pragma unroll
  for (int ms = 0; ms < 2; ++ms)
#pragma unroll
    for (int nt = 0; nt < 4; ++nt) {
      int col = (btile + wnt + nt) * 16 + mrow;
      float bb = bias ? bias[col] : 0.0f;
#pragma unroll
      for (int v = 0; v < 8; ++v) {
        int row = brow + wr + ms * 16 + hi * 8 + v;
        C[(size_t)row * ldc + col] = acc[ms][nt][v] + bb;
      }
    }
}

// ======================= init activations =======================
__global__ __launch_bounds__(256) void k_init_act(const float* __restrict__ T1,
                                                  const float* __restrict__ bq_b1,
                                                  const float* __restrict__ hi_b,
                                                  const float* __restrict__ ci_b,
                                                  _Float16* __restrict__ Gh,
                                                  float* __restrict__ HIDf,
                                                  _Float16* __restrict__ HIDh,
                                                  _Float16* __restrict__ A2h,
                                                  float* __restrict__ CELL) {
  int i = blockIdx.x * 256 + threadIdx.x;   // over B*H
  if (i >= BB * HH) return;
  int b = i >> 9, h = i & 511;
  const float* row = T1 + (size_t)b * 1536;
  float x = row[h] + bq_b1[h];
  float g = 0.5f * x * (1.0f + erff(x * 0.70710678118654752f));   // exact gelu
  Gh[i] = (_Float16)g;
  float hv = tanhf(row[512 + h] + hi_b[h]);
  HIDf[i] = hv;
  HIDh[i] = (_Float16)hv;
  A2h[(size_t)b * 1024 + 512 + h] = (_Float16)hv;
  CELL[i] = tanhf(row[1024 + h] + ci_b[h]);
}

// ======================= per-step: build x (and prev value) ====================
__global__ __launch_bounds__(256) void k_prep_x(const float* __restrict__ BQ,
                                                const float* __restrict__ step_emb,
                                                const float* __restrict__ emb,
                                                const float* __restrict__ pc_w,
                                                const float* __restrict__ pc_b,
                                                const float* __restrict__ eof_value,
                                                const int* __restrict__ CHOSEN,
                                                _Float16* __restrict__ A2h, int s) {
  int b = blockIdx.x, t = threadIdx.x;
  __shared__ float sEmb[DD];
  float p0 = 0.0f, p1 = 0.0f;                 // prev[h=t], prev[h=t+256]
  if (s > 0) {
    int c = CHOSEN[b];                        // uniform per block
    if (c == EOFI) {
      p0 = eof_value[t];
      p1 = eof_value[t + 256];
    } else {
      sEmb[t] = emb[((size_t)b * EE + c) * DD + t];
      __syncthreads();
      float a0 = pc_b[t], a1 = pc_b[t + 256];
#pragma unroll 4
      for (int d = 0; d < DD; ++d) {
        float ev = sEmb[d];
        a0 = fmaf(ev, pc_w[(size_t)d * HH + t], a0);
        a1 = fmaf(ev, pc_w[(size_t)d * HH + t + 256], a1);
      }
      p0 = a0; p1 = a1;
    }
  }
  const float* bq = BQ + (size_t)b * HH;
  const float* se = step_emb + (size_t)s * HH;
  A2h[(size_t)b * 1024 + t]       = (_Float16)(bq[t] + se[t] + p0);
  A2h[(size_t)b * 1024 + t + 256] = (_Float16)(bq[t + 256] + se[t + 256] + p1);
}

// ======================= per-step: LSTM pointwise =======================
__global__ __launch_bounds__(256) void k_lstm_gate(const float* __restrict__ GATES,
                                                   const float* __restrict__ b_ih,
                                                   const float* __restrict__ b_hh,
                                                   float* __restrict__ CELL,
                                                   float* __restrict__ HIDf,
                                                   _Float16* __restrict__ HIDh,
                                                   _Float16* __restrict__ A2h) {
  int i = blockIdx.x * 256 + threadIdx.x;     // over B*H
  if (i >= BB * HH) return;
  int b = i >> 9, h = i & 511;
  const float* g = GATES + (size_t)b * 2048;
  float iv = g[h]        + b_ih[h]        + b_hh[h];
  float fv = g[512 + h]  + b_ih[512 + h]  + b_hh[512 + h];
  float gv = g[1024 + h] + b_ih[1024 + h] + b_hh[1024 + h];
  float ov = g[1536 + h] + b_ih[1536 + h] + b_hh[1536 + h];
  float c = sigf(fv) * CELL[i] + sigf(iv) * tanhf(gv);
  CELL[i] = c;
  float hd = sigf(ov) * tanhf(c);
  HIDf[i] = hd;
  HIDh[i] = (_Float16)hd;
  A2h[(size_t)b * 1024 + 512 + h] = (_Float16)hd;
}

// ======================= per-step: attention logits + argmax + update ==========
// raw[b,e] = (q[b]·emb[b,e] + hidden·ep_b) / sqrt(H);  raw_eof = hidden·eof_key / sqrt(H)
__global__ __launch_bounds__(256) void k_attn_argmax(const float* __restrict__ Q,
                                                     const float* __restrict__ HIDf,
                                                     const float* __restrict__ emb,
                                                     const float* __restrict__ ep_b,
                                                     const float* __restrict__ eof_key,
                                                     int* __restrict__ REM,
                                                     int* __restrict__ FIN,
                                                     int* __restrict__ CHOSEN,
                                                     int* __restrict__ IDS,
                                                     int* __restrict__ SEL,
                                                     float* __restrict__ out, int s) {
  int b = blockIdx.x, t = threadIdx.x;
  __shared__ float sQ[DD];
  __shared__ float red[256];
  __shared__ float sv[256];
  __shared__ int   si[256];
  __shared__ float s_cb, s_reof;

  sQ[t] = Q[(size_t)b * DD + t];
  const float* hrow = HIDf + (size_t)b * HH;

  // hidden·ep_b
  red[t] = hrow[t] * ep_b[t] + hrow[t + 256] * ep_b[t + 256];
  __syncthreads();
  for (int o = 128; o > 0; o >>= 1) { if (t < o) red[t] += red[t + o]; __syncthreads(); }
  if (t == 0) s_cb = red[0];
  __syncthreads();
  // hidden·eof_key
  red[t] = hrow[t] * eof_key[t] + hrow[t + 256] * eof_key[t + 256];
  __syncthreads();
  for (int o = 128; o > 0; o >>= 1) { if (t < o) red[t] += red[t + o]; __syncthreads(); }
  if (t == 0) s_reof = red[0];
  __syncthreads();

  const int fin = FIN[b];
  int allow = 1;
  if (s == 0) {                               // allow_eof = !remaining.any()
    si[t] = REM[b * EE + t] | REM[b * EE + t + 256];
    __syncthreads();
    for (int o = 128; o > 0; o >>= 1) { if (t < o) si[t] |= si[t + o]; __syncthreads(); }
    allow = !si[0];
    __syncthreads();
  }

  const float cbv = s_cb;
  const float inv_scale = 0.044194173824159216f;   // 1/sqrt(512)

  __builtin_prefetch(emb + ((size_t)b * EE + t + 256) * DD, 0, 1);

  auto entLogit = [&](int e) -> float {
    if (fin || !REM[b * EE + e]) return NEGV;
    const float* er = emb + ((size_t)b * EE + e) * DD;
    float a = 0.0f;
#pragma unroll 4
    for (int d = 0; d < DD; d += 4) {
      a = fmaf(er[d],     sQ[d],     a);
      a = fmaf(er[d + 1], sQ[d + 1], a);
      a = fmaf(er[d + 2], sQ[d + 2], a);
      a = fmaf(er[d + 3], sQ[d + 3], a);
    }
    return (a + cbv) * inv_scale;
  };

  float l0 = entLogit(t);
  float l1 = entLogit(t + 256);
  size_t obase = ((size_t)b * SS + s) * (EE + 1);
  out[obase + t]       = l0;
  out[obase + t + 256] = l1;

  float bestv = l0; int besti = t;
  if (l1 > bestv) { bestv = l1; besti = t + 256; }
  if (t == 0) {
    float le = (fin || allow) ? s_reof * inv_scale : NEGV;
    out[obase + EE] = le;
    if (le > bestv) { bestv = le; besti = EOFI; }
  }
  sv[t] = bestv; si[t] = besti;
  __syncthreads();
  for (int o = 128; o > 0; o >>= 1) {
    if (t < o) {
      float v2 = sv[t + o]; int i2 = si[t + o];
      if (v2 > sv[t] || (v2 == sv[t] && i2 < si[t])) { sv[t] = v2; si[t] = i2; }
    }
    __syncthreads();
  }
  if (t == 0) {
    int ch = si[0];
    int non_eof = (!fin) && (ch != EOFI);
    if (non_eof) REM[b * EE + ch] = 0;
    FIN[b] = fin || (ch == EOFI);
    CHOSEN[b] = ch;
    IDS[b * SS + s] = ch;
    SEL[b * SS + s] = non_eof;
    out[IDS_OFF + (size_t)b * SS + s] = (float)ch;
    out[SEL_OFF + (size_t)b * SS + s] = (float)non_eof;
  }
}

// ======================= summary: masked mean of selected embeddings ===========
__global__ __launch_bounds__(256) void k_summarize(const float* __restrict__ emb,
                                                   const int* __restrict__ IDS,
                                                   const int* __restrict__ SEL,
                                                   float* __restrict__ out) {
  int b = blockIdx.x, d = threadIdx.x;          // d in [0,256)
  float acc = 0.0f; int cnt = 0;
#pragma unroll
  for (int s = 0; s < SS; ++s) {
    if (SEL[b * SS + s]) {
      ++cnt;
      acc += emb[((size_t)b * EE + IDS[b * SS + s]) * DD + d];
    }
  }
  out[SUM_OFF + (size_t)b * DD + d] = acc / fmaxf((float)cnt, 1.0f);
}

// ======================= host launcher =======================
extern "C" void kernel_launch(void* const* d_in, const int* in_sizes, int n_in,
                              void* d_out, int out_size, void* d_ws, size_t ws_size,
                              hipStream_t stream) {
  (void)in_sizes; (void)n_in; (void)out_size; (void)ws_size;
  const float* latent   = (const float*)d_in[0];
  const float* emb      = (const float*)d_in[1];
  const int*   mask     = (const int*)  d_in[2];
  const float* step_emb = (const float*)d_in[3];
  const float* bq_w1    = (const float*)d_in[4];
  const float* bq_b1    = (const float*)d_in[5];
  const float* bq_w2    = (const float*)d_in[6];
  const float* bq_b2    = (const float*)d_in[7];
  const float* hi_w     = (const float*)d_in[8];
  const float* hi_b     = (const float*)d_in[9];
  const float* ci_w     = (const float*)d_in[10];
  const float* ci_b     = (const float*)d_in[11];
  const float* w_ih     = (const float*)d_in[12];
  const float* b_ih     = (const float*)d_in[13];
  const float* w_hh     = (const float*)d_in[14];
  const float* b_hh     = (const float*)d_in[15];
  const float* ep_w     = (const float*)d_in[16];
  const float* ep_b     = (const float*)d_in[17];
  const float* pc_w     = (const float*)d_in[18];
  const float* pc_b     = (const float*)d_in[19];
  const float* eof_key  = (const float*)d_in[20];
  const float* eof_val  = (const float*)d_in[21];
  float* out = (float*)d_out;
  char*  ws  = (char*)d_ws;

  _Float16* W2p   = (_Float16*)(ws + OFF_W2P);
  _Float16* WC1p  = (_Float16*)(ws + OFF_WC1P);
  _Float16* BQ2p  = (_Float16*)(ws + OFF_BQ2P);
  _Float16* EPWp  = (_Float16*)(ws + OFF_EPWP);
  _Float16* LATh  = (_Float16*)(ws + OFF_LATH);
  _Float16* A2h   = (_Float16*)(ws + OFF_A2H);
  _Float16* Gh    = (_Float16*)(ws + OFF_GH);
  _Float16* HIDh  = (_Float16*)(ws + OFF_HIDH);
  float*    T1    = (float*)(ws + OFF_T1);
  float*    GATES = (float*)(ws + OFF_GATES);
  float*    BQf   = (float*)(ws + OFF_BQ);
  float*    HIDf  = (float*)(ws + OFF_HIDF);
  float*    CELL  = (float*)(ws + OFF_CELL);
  float*    Qf    = (float*)(ws + OFF_Q);
  int*      REM   = (int*)(ws + OFF_REM);
  int*      FIN   = (int*)(ws + OFF_FIN);
  int*      CHO   = (int*)(ws + OFF_CHO);
  int*      IDS   = (int*)(ws + OFF_IDS);
  int*      SEL   = (int*)(ws + OFF_SEL);

  // ---- one-time packing (re-done every call: deterministic) ----
  k_pack_w2_frag  <<<2048, 256, 0, stream>>>(w_ih, w_hh, W2p);
  k_pack_wc1_frag <<<2048, 256, 0, stream>>>(bq_w1, hi_w, ci_w, WC1p);
  k_pack_bq2_frag <<<1024, 256, 0, stream>>>(bq_w2, BQ2p);
  k_pack_epwt_frag<<<512, 256, 0, stream>>>(ep_w, EPWp);
  k_cvt_f16       <<<2048, 256, 0, stream>>>(latent, LATh, BB * LL);
  k_init_state    <<<1024, 256, 0, stream>>>(mask, REM, FIN);

  // ---- init projections: T1 = latent @ [bq_w1|hi_w|ci_w] ----
  k_gemm_wmma<<<dim3(1536 / 128, BB / 64), 128, 0, stream>>>(
      LATh, WC1p, nullptr, T1, 1536, LL, LL, 1536);
  k_init_act<<<1024, 256, 0, stream>>>(T1, bq_b1, hi_b, ci_b, Gh, HIDf, HIDh, A2h, CELL);
  // base_query = gelu(...) @ bq_w2 + bq_b2
  k_gemm_wmma<<<dim3(HH / 128, BB / 64), 128, 0, stream>>>(
      Gh, BQ2p, bq_b2, BQf, HH, HH, HH, HH);

  // ---- 13 sequential decode steps ----
  for (int s = 0; s < SS; ++s) {
    k_prep_x<<<BB, 256, 0, stream>>>(BQf, step_emb, emb, pc_w, pc_b, eof_val, CHO, A2h, s);
    // gates = [x|hidden] @ [w_ih;w_hh]   (biases added in pointwise kernel)
    k_gemm_wmma<<<dim3(2048 / 128, BB / 64), 128, 0, stream>>>(
        A2h, W2p, nullptr, GATES, 2048, 1024, 1024, 2048);
    k_lstm_gate<<<1024, 256, 0, stream>>>(GATES, b_ih, b_hh, CELL, HIDf, HIDh, A2h);
    // q = hidden @ ep_w^T   (down-projection trick, exact reassociation)
    k_gemm_wmma<<<dim3(DD / 128, BB / 64), 128, 0, stream>>>(
        HIDh, EPWp, nullptr, Qf, DD, HH, HH, DD);
    k_attn_argmax<<<BB, 256, 0, stream>>>(Qf, HIDf, emb, ep_b, eof_key,
                                          REM, FIN, CHO, IDS, SEL, out, s);
  }

  // ---- summary ----
  k_summarize<<<BB, 256, 0, stream>>>(emb, IDS, SEL, out);
}
